// Attention_9363028705744
// MI455X (gfx1250) — matrix operand order repaired and checked by reading, not compiled, not run on hardware
//
#include <hip/hip_runtime.h>
#include <hip/hip_bf16.h>

// -------- problem constants --------
#define BB 4
#define SS 2048
#define DD 512
#define HH 8
#define HD 64
#define NT (BB * SS)          // 8192 tokens
#define PITCH (SS + 8)        // f32 LDS score-slab pitch (floats)
#define P16   (SS + 16)       // f16 LDS prob-slab pitch (halves), 32B aligned

typedef _Float16 half_t;
typedef __attribute__((ext_vector_type(16))) _Float16 v16h;
typedef __attribute__((ext_vector_type(8)))  float    v8f;

// ================= WMMA fragment helpers (layouts per CDNA5 ISA 7.12.2) ====

__device__ __forceinline__ v8f wmma_f16(v16h a, v16h b, v8f c) {
  // (neg_a, A, neg_b, B, c_mod, C, reuse_a, reuse_b)
  return __builtin_amdgcn_wmma_f32_16x16x32_f16(false, a, false, b, (short)0, c,
                                                false, false);
}

// A fragment 16x32 f16: lanes 0-15 -> M=lane, K = {0..7, 16..23}+k0
//                       lanes 16-31 -> M=lane-16, K = {8..15, 24..31}+k0
__device__ __forceinline__ v16h frag_a_f32(const float* __restrict__ p, int ld,
                                           int m0, int k0, int lane) {
  const int row = m0 + (lane & 15);
  const int kb  = k0 + ((lane >> 4) << 3);
  const float* s = p + (size_t)row * ld + kb;
  v16h a;
#pragma unroll
  for (int e = 0; e < 8; ++e) {
    a[e]     = (half_t)s[e];
    a[e + 8] = (half_t)s[16 + e];
  }
  return a;
}

__device__ __forceinline__ v16h frag_a_f16(const half_t* __restrict__ p, int ld,
                                           int m0, int k0, int lane) {
  const int row = m0 + (lane & 15);
  const int kb  = k0 + ((lane >> 4) << 3);
  const half_t* s = p + (size_t)row * ld + kb;
  v16h a;
#pragma unroll
  for (int e = 0; e < 8; ++e) {
    a[e]     = s[e];
    a[e + 8] = s[16 + e];
  }
  return a;
}

// B fragment 32x16 where memory holds the transpose: B(k,n) = p[n*ld + k].
// lane%16 = N; lanes 0-15 hold K = {0..7, 16..23}+k0,
// lanes 16-31 hold K = {8..15, 24..31}+k0 (matches A fragment).
__device__ __forceinline__ v16h frag_bt_f16(const half_t* __restrict__ p, int ld,
                                            int k0, int n0, int lane) {
  const int row = n0 + (lane & 15);
  const int kb  = k0 + ((lane >> 4) << 3);
  const half_t* s = p + (size_t)row * ld + kb;
  v16h b;
#pragma unroll
  for (int e = 0; e < 8; ++e) {
    b[e]     = s[e];
    b[e + 8] = s[16 + e];
  }
  return b;
}

// ================= Kernel 0: weight convert + transpose ====================
// W [512,512] f32 row-major -> WT [n][k] f16. grid (16,16,4), block 256.
__global__ __launch_bounds__(256) void wcvt_kernel(
    const float* __restrict__ W0, const float* __restrict__ W1,
    const float* __restrict__ W2, const float* __restrict__ W3,
    half_t* __restrict__ T0, half_t* __restrict__ T1,
    half_t* __restrict__ T2, half_t* __restrict__ T3) {
  __shared__ float tile[32][33];
  const int z = blockIdx.z;
  const float* W = (z == 0) ? W0 : (z == 1) ? W1 : (z == 2) ? W2 : W3;
  half_t*      T = (z == 0) ? T0 : (z == 1) ? T1 : (z == 2) ? T2 : T3;
  const int tx = threadIdx.x & 31;
  const int ty = threadIdx.x >> 5;
  const int bx = blockIdx.x * 32, by = blockIdx.y * 32;
#pragma unroll
  for (int i = ty; i < 32; i += 8)
    tile[i][tx] = W[(size_t)(by + i) * DD + bx + tx];
  __syncthreads();
#pragma unroll
  for (int i = ty; i < 32; i += 8)
    T[(size_t)(bx + i) * DD + by + tx] = (half_t)tile[tx][i];
}

// ================= Kernel 1: fused QKV projection ==========================
// grid (NT/16 = 512, 3), block 256 (8 waves). Wave w -> 16x64 tile (n0=w*64),
// 4 accumulators; ping-pong double-buffered fragment loads (2 k-steps/iter).
__global__ __launch_bounds__(256) void qkv_proj_kernel(
    const float* __restrict__ X,
    const float* __restrict__ bq, const float* __restrict__ bk,
    const float* __restrict__ bv,
    const half_t* __restrict__ WqT, const half_t* __restrict__ WkT,
    const half_t* __restrict__ WvT,
    half_t* __restrict__ Qh, half_t* __restrict__ Kh,
    half_t* __restrict__ Vh, half_t* __restrict__ VhT) {
  const int lane = threadIdx.x & 31;
  const int wave = threadIdx.x >> 5;
  const int which = blockIdx.y;                     // 0=Q 1=K 2=V
  const int m0 = blockIdx.x * 16;                   // token tile
  const int n0 = wave * 64;                         // 8 waves cover all 512 cols

  const half_t* WT   = (which == 0) ? WqT : (which == 1) ? WkT : WvT;
  const float*  bias = (which == 0) ? bq  : (which == 1) ? bk  : bv;
  half_t*       dst  = (which == 0) ? Qh  : (which == 1) ? Kh  : Vh;

  v8f c[4] = {{}, {}, {}, {}};
  v16h a0 = frag_a_f32(X, DD, m0, 0, lane);
  v16h b0[4];
#pragma unroll
  for (int t = 0; t < 4; ++t) b0[t] = frag_bt_f16(WT, DD, 0, n0 + 16 * t, lane);

  for (int k0 = 0; k0 < DD; k0 += 64) {             // 2 k-steps per iteration
    v16h a1 = frag_a_f32(X, DD, m0, k0 + 32, lane);
    v16h b1[4];
#pragma unroll
    for (int t = 0; t < 4; ++t)
      b1[t] = frag_bt_f16(WT, DD, k0 + 32, n0 + 16 * t, lane);
#pragma unroll
    for (int t = 0; t < 4; ++t) c[t] = wmma_f16(a0, b0[t], c[t]);

    const int kn = (k0 + 64 < DD) ? k0 + 64 : 0;    // clamped wrap prefetch
    a0 = frag_a_f32(X, DD, m0, kn, lane);
#pragma unroll
    for (int t = 0; t < 4; ++t)
      b0[t] = frag_bt_f16(WT, DD, kn, n0 + 16 * t, lane);
#pragma unroll
    for (int t = 0; t < 4; ++t) c[t] = wmma_f16(a1, b1[t], c[t]);
  }

  const int rb = m0 + ((lane >> 4) << 3);
#pragma unroll
  for (int t = 0; t < 4; ++t) {
    const int col = n0 + 16 * t + (lane & 15);
    const float bs = bias[col];
    const int h = col >> 6, hd = col & (HD - 1);
#pragma unroll
    for (int v = 0; v < 8; ++v) {
      const int tok = rb + v;
      const int b_ = tok >> 11, s_ = tok & (SS - 1);
      const half_t val = (half_t)(c[t][v] + bs);
      dst[(((size_t)b_ * HH + h) * SS + s_) * HD + hd] = val;
      if (which == 2)   // V^T: [b,h,hd,key] for contiguous B-frags in attn@V
        VhT[(((size_t)b_ * HH + h) * HD + hd) * SS + s_] = val;
    }
  }
}

// ================= Kernel 2: fused scores + softmax + attn write + ctx =====
// grid (S/16 = 128, B*H = 32), block 256 (8 waves).
// Dynamic LDS: 16xPITCH f32 scores + 4x32x8 f32 reduction + 16xP16 f16 probs.
__global__ __launch_bounds__(256) void attn_fused_kernel(
    const half_t* __restrict__ Qh, const half_t* __restrict__ Kh,
    const half_t* __restrict__ VhT, const float* __restrict__ mask,
    float* __restrict__ attn_out, half_t* __restrict__ ctx) {
  extern __shared__ float sc[];                     // [16][PITCH] f32
  float*  red = sc + (size_t)16 * PITCH;            // [4][32][8]  f32
  half_t* p16 = (half_t*)(red + 4 * 32 * 8);        // [16][P16]   f16
  const int lane = threadIdx.x & 31;
  const int wave = threadIdx.x >> 5;
  const int q0 = blockIdx.x * 16;
  const int bh = blockIdx.y;
  const int b  = bh >> 3;

  const half_t* Qp  = Qh  + (size_t)bh * SS * HD;
  const half_t* Kp  = Kh  + (size_t)bh * SS * HD;
  const half_t* VpT = VhT + (size_t)bh * HD * SS;

  // Pull the V^T slab (256 KB, needed in phase 3b) toward the WGP while the
  // score WMMAs run: lowers to global_prefetch_b8.
  for (size_t off = (size_t)threadIdx.x * 128; off < (size_t)HD * SS;
       off += (size_t)256 * 128)
    __builtin_prefetch(VpT + off, 0, 1);

  // ---- phase 1: scores tile row (16 x 2048) into LDS, ping-pong ----
  const v16h aq0 = frag_a_f16(Qp, HD, q0, 0, lane);
  const v16h aq1 = frag_a_f16(Qp, HD, q0, 32, lane);

  auto store_tile = [&](int kt, const v8f& c) {
    const int col = kt * 16 + (lane & 15);
    const float mterm = -1.0e6f * (1.0f - mask[b * SS + col]);
    const int rb = (lane >> 4) << 3;
#pragma unroll
    for (int v = 0; v < 8; ++v)
      sc[(size_t)(rb + v) * PITCH + col] = c[v] * 0.125f + mterm;  // 1/sqrt(64)
  };

  v16h bka0 = frag_bt_f16(Kp, HD, 0, wave * 16, lane);
  v16h bka1 = frag_bt_f16(Kp, HD, 32, wave * 16, lane);
  for (int kt = wave; kt < SS / 16; kt += 16) {     // handles kt and kt+8
    v16h bkb0 = frag_bt_f16(Kp, HD, 0, (kt + 8) * 16, lane);
    v16h bkb1 = frag_bt_f16(Kp, HD, 32, (kt + 8) * 16, lane);
    {
      v8f c = {};
      c = wmma_f16(aq0, bka0, c);
      c = wmma_f16(aq1, bka1, c);
      store_tile(kt, c);
    }
    const int ktn = (kt + 16 < SS / 16) ? kt + 16 : kt;
    bka0 = frag_bt_f16(Kp, HD, 0, ktn * 16, lane);
    bka1 = frag_bt_f16(Kp, HD, 32, ktn * 16, lane);
    {
      v8f c = {};
      c = wmma_f16(aq0, bkb0, c);
      c = wmma_f16(aq1, bkb1, c);
      store_tile(kt + 8, c);
    }
  }
  __syncthreads();

  // ---- phase 2: row softmax, 16 threads per row (wave32 shuffle reduce) ----
  // Writes normalized probs both as f32 (for attn output) and f16 (for WMMA).
  {
    const int r = threadIdx.x >> 4;
    const int c16 = threadIdx.x & 15;
    float*  row   = sc  + (size_t)r * PITCH;
    half_t* row16 = p16 + (size_t)r * P16;
    float mx = -3.4e38f;
    for (int j = c16; j < SS; j += 16) mx = fmaxf(mx, row[j]);
#pragma unroll
    for (int off = 8; off; off >>= 1) mx = fmaxf(mx, __shfl_xor(mx, off, 16));
    float sum = 0.0f;
    for (int j = c16; j < SS; j += 16) {
      float e = __expf(row[j] - mx);
      row[j] = e;
      sum += e;
    }
#pragma unroll
    for (int off = 8; off; off >>= 1) sum += __shfl_xor(sum, off, 16);
    const float inv = 1.0f / sum;
    for (int j = c16; j < SS; j += 16) {
      const float p = row[j] * inv;
      row[j]   = p;
      row16[j] = (half_t)p;
    }
  }
  __syncthreads();

  // ---- phase 3a: coalesced float4 write of probabilities to d_out ----
  {
    float* aout = attn_out + ((size_t)bh * SS + q0) * SS;
    for (int idx = threadIdx.x; idx < 16 * (SS / 4); idx += 256) {
      const int rr = idx / (SS / 4);
      const int jj = (idx % (SS / 4)) << 2;
      float4 v4 = *(const float4*)(sc + (size_t)rr * PITCH + jj);
      *(float4*)(aout + (size_t)rr * SS + jj) = v4;
    }
  }

  // ---- phase 3b: ctx = attn @ V from f16 LDS; 8 waves, K split in half ----
  const int nt = (wave & 3) * 16;                   // hd column tile
  const int kb = (wave >> 2) * (SS / 2);            // key-range half
  v8f c = {};
  v16h pa0 = frag_a_f16(p16, P16, 0, kb, lane);
  v16h vb0 = frag_bt_f16(VpT, SS, kb, nt, lane);
  for (int k0 = kb; k0 < kb + SS / 2; k0 += 64) {   // 2 k-steps per iteration
    v16h pa1 = frag_a_f16(p16, P16, 0, k0 + 32, lane);
    v16h vb1 = frag_bt_f16(VpT, SS, k0 + 32, nt, lane);
    c = wmma_f16(pa0, vb0, c);
    const int kn = (k0 + 64 < kb + SS / 2) ? k0 + 64 : kb;
    pa0 = frag_a_f16(p16, P16, 0, kn, lane);
    vb0 = frag_bt_f16(VpT, SS, kn, nt, lane);
    c = wmma_f16(pa1, vb1, c);
  }
  if (wave >= 4) {
#pragma unroll
    for (int v = 0; v < 8; ++v)
      red[(size_t)((wave - 4) * 32 + lane) * 8 + v] = c[v];
  }
  __syncthreads();
  if (wave < 4) {
#pragma unroll
    for (int v = 0; v < 8; ++v)
      c[v] += red[(size_t)(wave * 32 + lane) * 8 + v];
    const int h = bh & 7;
    const int col = h * HD + nt + (lane & 15);
    const int rb = (lane >> 4) << 3;
#pragma unroll
    for (int v = 0; v < 8; ++v) {
      const int s_ = q0 + rb + v;
      ctx[((size_t)b * SS + s_) * (HH * HD) + col] = (half_t)c[v];
    }
  }
}

// ================= Kernel 3: output projection =============================
// grid (NT/16 = 512), block 256 (8 waves). Wave w -> 16x64 tile, ping-pong.
__global__ __launch_bounds__(256) void out_proj_kernel(
    const half_t* __restrict__ ctx, const half_t* __restrict__ WoT,
    const float* __restrict__ bo, float* __restrict__ out) {
  const int lane = threadIdx.x & 31;
  const int wave = threadIdx.x >> 5;
  const int m0 = blockIdx.x * 16;
  const int n0 = wave * 64;

  v8f c[4] = {{}, {}, {}, {}};
  v16h a0 = frag_a_f16(ctx, HH * HD, m0, 0, lane);
  v16h b0[4];
#pragma unroll
  for (int t = 0; t < 4; ++t) b0[t] = frag_bt_f16(WoT, DD, 0, n0 + 16 * t, lane);

  for (int k0 = 0; k0 < HH * HD; k0 += 64) {        // 2 k-steps per iteration
    v16h a1 = frag_a_f16(ctx, HH * HD, m0, k0 + 32, lane);
    v16h b1[4];
#pragma unroll
    for (int t = 0; t < 4; ++t)
      b1[t] = frag_bt_f16(WoT, DD, k0 + 32, n0 + 16 * t, lane);
#pragma unroll
    for (int t = 0; t < 4; ++t) c[t] = wmma_f16(a0, b0[t], c[t]);

    const int kn = (k0 + 64 < HH * HD) ? k0 + 64 : 0;
    a0 = frag_a_f16(ctx, HH * HD, m0, kn, lane);
#pragma unroll
    for (int t = 0; t < 4; ++t)
      b0[t] = frag_bt_f16(WoT, DD, kn, n0 + 16 * t, lane);
#pragma unroll
    for (int t = 0; t < 4; ++t) c[t] = wmma_f16(a1, b1[t], c[t]);
  }

  const int rb = m0 + ((lane >> 4) << 3);
#pragma unroll
  for (int t = 0; t < 4; ++t) {
    const int col = n0 + 16 * t + (lane & 15);
    const float bs = bo[col];
#pragma unroll
    for (int v = 0; v < 8; ++v)
      out[(size_t)(rb + v) * DD + col] = c[t][v] + bs;
  }
}

// ================= host launch =============================================
extern "C" void kernel_launch(void* const* d_in, const int* in_sizes, int n_in,
                              void* d_out, int out_size, void* d_ws, size_t ws_size,
                              hipStream_t stream) {
  (void)in_sizes; (void)n_in; (void)out_size; (void)ws_size;
  const float* X    = (const float*)d_in[0];
  const float* mask = (const float*)d_in[1];
  const float* Wq   = (const float*)d_in[2];
  const float* bq   = (const float*)d_in[3];
  const float* Wk   = (const float*)d_in[4];
  const float* bk   = (const float*)d_in[5];
  const float* Wv   = (const float*)d_in[6];
  const float* bv   = (const float*)d_in[7];
  const float* Wo   = (const float*)d_in[8];
  const float* bo   = (const float*)d_in[9];

  float* out  = (float*)d_out;
  float* attn = out + (size_t)BB * SS * DD;         // outputs concatenated

  const size_t qkv_elems = (size_t)BB * HH * SS * HD;   // 4.19M halves
  const size_t w_elems   = (size_t)DD * DD;             // 262144 halves
  half_t* Qh  = (half_t*)d_ws;
  half_t* Kh  = Qh  + qkv_elems;
  half_t* Vh  = Kh  + qkv_elems;
  half_t* VhT = Vh  + qkv_elems;                    // [B,H,HD,S] f16
  half_t* ctx = VhT + qkv_elems;                    // [B,S,H*HD] f16
  half_t* WqT = ctx + qkv_elems;                    // [n][k] f16
  half_t* WkT = WqT + w_elems;
  half_t* WvT = WkT + w_elems;
  half_t* WoT = WvT + w_elems;

  wcvt_kernel<<<dim3(16, 16, 4), 256, 0, stream>>>(
      Wq, Wk, Wv, Wo, WqT, WkT, WvT, WoT);

  qkv_proj_kernel<<<dim3(NT / 16, 3), 256, 0, stream>>>(
      X, bq, bk, bv, WqT, WkT, WvT, Qh, Kh, Vh, VhT);

  const size_t lds_bytes =
      ((size_t)16 * PITCH + 4 * 32 * 8) * sizeof(float) +
      (size_t)16 * P16 * sizeof(half_t);            // ~197 KB of 320 KB WGP LDS
  attn_fused_kernel<<<dim3(SS / 16, BB * HH), 256, lds_bytes, stream>>>(
      Qh, Kh, VhT, mask, attn, ctx);

  out_proj_kernel<<<dim3(NT / 16), 256, 0, stream>>>(ctx, WoT, bo, out);
}
